// AttentionHead_16028817949042
// MI455X (gfx1250) — compile-verified, matrix operand order checked by reading, not run on hardware
//
#include <hip/hip_runtime.h>
#include <hip/hip_bf16.h>

// Attention: B=4, S=4096, D=1024, DK=DV=64.
// d_out = output [B,S,64] (f32) followed by weights [B,S,S] (f32), flat.
// d_ws  = Qf (row-major f16) | Kf (row-major f16) | Vt ([B][64][S] f16)
//         | Wt ([3][64][1024] f16, transposed weights)

#define Bn 4
#define Sn 4096
#define Dn 1024
#define DKn 64
#define NTOK (Bn * Sn)            // 16384 token rows
#define QTILES (Sn / 16)          // 256 query/key 16-tiles per sequence

typedef __attribute__((ext_vector_type(16))) _Float16 v16h;
typedef __attribute__((ext_vector_type(8)))  _Float16 v8h;
typedef __attribute__((ext_vector_type(4)))  _Float16 v4h;
typedef __attribute__((ext_vector_type(8)))  float    v8f;
typedef __attribute__((ext_vector_type(4)))  float    v4f;   // clang vector: OK for NT builtins

// ---------------------------------------------------------------------------
// Kernel 0: transpose + convert projection weights to f16.
// Wt[m][col][k] = W_m[k][col].  3*64*1024 = 196608 elements (one-time, tiny).
// ---------------------------------------------------------------------------
__global__ void wprep_kernel(const float* __restrict__ Wq,
                             const float* __restrict__ Wk,
                             const float* __restrict__ Wv,
                             _Float16* __restrict__ Wt) {
  const int idx = blockIdx.x * 256 + threadIdx.x;     // < 196608
  const int m   = idx >> 16;                          // 0..2
  const int rem = idx & 0xFFFF;
  const int col = rem >> 10;                          // 0..63
  const int k   = rem & 0x3FF;                        // 0..1023
  const float* W = (m == 0) ? Wq : (m == 1) ? Wk : Wv;
  Wt[idx] = (_Float16)W[(size_t)k * DKn + col];
}

// ---------------------------------------------------------------------------
// Kernel 1: fused QKV projection.  One 4-wave block per 16-token row tile.
// Each k-step stages the 16x32 x strip in LDS *as f16* (one coalesced b128
// global load + one cvt + one ds_store_b64 per thread -> x read from HBM
// exactly once, converted exactly once).  Wave w owns output columns
// [16w,16w+16) of Q, K and V (3 WMMAs per k-step per wave, B frags from Wt).
// grid = NTOK/16, block = 128.
// ---------------------------------------------------------------------------
__global__ void qkv_proj_kernel(const float* __restrict__ x,
                                const _Float16* __restrict__ Wt,
                                const float* __restrict__ bq,
                                const float* __restrict__ bk,
                                const float* __restrict__ bv,
                                _Float16* __restrict__ Qf,
                                _Float16* __restrict__ Kf,
                                _Float16* __restrict__ Vt) {
  __shared__ _Float16 xs[16][32];        // current x strip, f16 (1 KB)

  const int tile = blockIdx.x;           // 16-token row tile
  const int lane = threadIdx.x & 31;
  const int wave = threadIdx.x >> 5;     // 0..3 = output column tile
  const int ln   = lane & 15;
  const int hi   = lane >> 4;

  // cooperative x-strip load mapping: one 16B load per thread per k-step
  const int ldRow = threadIdx.x >> 3;          // 0..15
  const int ldCol = (threadIdx.x & 7) * 4;     // 0,4,..,28
  const float* xrow = x + (size_t)(16 * tile + ldRow) * Dn;

  const int col = 16 * wave + ln;              // output column held by lane
  const _Float16* wtq = Wt + 0 * 65536 + (size_t)col * Dn;
  const _Float16* wtk = Wt + 1 * 65536 + (size_t)col * Dn;
  const _Float16* wtv = Wt + 2 * 65536 + (size_t)col * Dn;

  v8f accQ = {}, accK = {}, accV = {};
  for (int k0 = 0; k0 < Dn; k0 += 32) {
    __syncthreads();
    {
      const v4f t = *(const v4f*)(xrow + k0 + ldCol);
      v4h h;
      h[0] = (_Float16)t[0]; h[1] = (_Float16)t[1];
      h[2] = (_Float16)t[2]; h[3] = (_Float16)t[3];
      *(v4h*)&xs[ldRow][ldCol] = h;            // convert once, share 4 ways
    }
    if (k0 + 32 < Dn)                          // prefetch next strip near-cache
      __builtin_prefetch(xrow + k0 + 32 + ldCol, 0, 3);
    __syncthreads();

    // A fragment (16x32 f16): lane = row ln, halfs hi*8 + {0..7, 16..23}
    // -> two contiguous 16B ds loads per lane
    v16h a;
    const _Float16* xp = &xs[ln][hi * 8];
#pragma unroll
    for (int i = 0; i < 8; ++i) { a[i] = xp[i]; a[8 + i] = xp[16 + i]; }

    // B fragments from transposed weights: two contiguous 16B runs per lane
    v16h bQ, bK, bV;
    const _Float16* q = wtq + k0 + hi * 8;
    const _Float16* k = wtk + k0 + hi * 8;
    const _Float16* v = wtv + k0 + hi * 8;
#pragma unroll
    for (int i = 0; i < 8; ++i) {
      bQ[i] = q[i]; bQ[8 + i] = q[16 + i];
      bK[i] = k[i]; bK[8 + i] = k[16 + i];
      bV[i] = v[i]; bV[8 + i] = v[16 + i];
    }
    accQ = __builtin_amdgcn_wmma_f32_16x16x32_f16(false, a, false, bQ,
                                                  (short)0, accQ, false, false);
    accK = __builtin_amdgcn_wmma_f32_16x16x32_f16(false, a, false, bK,
                                                  (short)0, accK, false, false);
    accV = __builtin_amdgcn_wmma_f32_16x16x32_f16(false, a, false, bV,
                                                  (short)0, accV, false, false);
  }

  const float bbq = bq[col], bbk = bk[col], bbv = bv[col];
  // Q, K: row-major f16 [B*S, 64]
#pragma unroll
  for (int r = 0; r < 8; ++r) {
    const int row = 16 * tile + r + 8 * hi;     // C layout: M = r + 8*hi, N = ln
    Qf[(size_t)row * DKn + col] = (_Float16)(accQ[r] + bbq);
    Kf[(size_t)row * DKn + col] = (_Float16)(accK[r] + bbk);
  }
  // V: transposed [B][64][S]; lane's 8 rows are consecutive in S -> one b128
  {
    const int row0 = 16 * tile + 8 * hi;
    const int b    = row0 >> 12;                // 16-row tiles never straddle B
    const int s0   = row0 & (Sn - 1);
    v8h vv;
#pragma unroll
    for (int r = 0; r < 8; ++r) vv[r] = (_Float16)(accV[r] + bbv);
    *(v8h*)(Vt + ((size_t)b * DKn + col) * Sn + s0) = vv;
  }
}

// ---------------------------------------------------------------------------
// Kernel 2: fused causal attention for one 16-row query strip.
// grid = (S/16, B), block = 128 (4 waves), dynLDS = 16*4096*4 = 256 KB.
//   phase 1: scores (WMMA, causal) -> LDS strip [16 x 4096] f32
//   phase 2: softmax in LDS (v4f lanes); weights streamed NT to global
//   phase 3: output = P @ V (WMMA over causal K range, V from Vt contiguous)
// ---------------------------------------------------------------------------
__global__ void attn_kernel(const _Float16* __restrict__ Qf,
                            const _Float16* __restrict__ Kf,
                            const _Float16* __restrict__ Vt,
                            float* __restrict__ out,     // [B,S,64]
                            float* __restrict__ wout) {  // [B,S,S]
  extern __shared__ float sc[];          // [16][4096] score strip

  const int b     = blockIdx.y;
  const int qt    = blockIdx.x;          // query 16-tile index
  const int qBase = qt * 16;
  const int lane  = threadIdx.x & 31;
  const int wave  = threadIdx.x >> 5;    // 0..3
  const int ln    = lane & 15;
  const int hi    = lane >> 4;

  const _Float16* Qb = Qf + (size_t)b * Sn * DKn;
  const _Float16* Kb = Kf + (size_t)b * Sn * DKn;
  const _Float16* Vb = Vt + (size_t)b * DKn * Sn;

  const int cEnd   = qBase + 16;                       // real scores in [0,cEnd)
  int padEnd = (cEnd + 127) & ~127;                    // 128-align for v4f loops
  if (padEnd > Sn) padEnd = Sn;

  // ---- Q fragments for this strip (K=64 -> two 16x32 A fragments) ----
  v16h qa0, qa1;
  {
    const _Float16* qp = Qb + (size_t)(qBase + ln) * DKn + hi * 8;
#pragma unroll
    for (int i = 0; i < 8; ++i) { qa0[i] = qp[i];      qa0[8 + i] = qp[16 + i]; }
#pragma unroll
    for (int i = 0; i < 8; ++i) { qa1[i] = qp[32 + i]; qa1[8 + i] = qp[48 + i]; }
  }

  // ---- phase 1: score strip into LDS (causal tiles + -inf pad tiles) ----
  const int jLast = (padEnd >> 4) - 1;
  for (int j = wave; j <= jLast; j += 4) {
    if (j > qt) {                         // fully-masked pad tile
#pragma unroll
      for (int r = 0; r < 8; ++r)
        sc[(r + 8 * hi) * Sn + j * 16 + ln] = -1e30f;
      continue;
    }
    const int kBase = j * 16;
    // K^T as B fragments: lane holds key row kBase+ln, d = hi*8+{0..7,16..23}
    const _Float16* kp = Kb + (size_t)(kBase + ln) * DKn + hi * 8;
    v16h kb0, kb1;
#pragma unroll
    for (int i = 0; i < 8; ++i) { kb0[i] = kp[i];      kb0[8 + i] = kp[16 + i]; }
#pragma unroll
    for (int i = 0; i < 8; ++i) { kb1[i] = kp[32 + i]; kb1[8 + i] = kp[48 + i]; }

    v8f s = {};
    s = __builtin_amdgcn_wmma_f32_16x16x32_f16(false, qa0, false, kb0,
                                               (short)0, s, false, false);
    s = __builtin_amdgcn_wmma_f32_16x16x32_f16(false, qa1, false, kb1,
                                               (short)0, s, false, false);
#pragma unroll
    for (int r = 0; r < 8; ++r) {
      const int m = r + 8 * hi;           // query row within strip
      float v = s[r] * 0.125f;            // 1/sqrt(64)
      if (j == qt && ln > m) v = -1e30f;  // causal mask on diagonal tile
      sc[m * Sn + kBase + ln] = v;
    }
  }
  __syncthreads();

  // ---- phase 2: softmax per row (v4f lanes); stream weights NT ----
  for (int rr = 0; rr < 4; ++rr) {
    const int m = wave * 4 + rr;
    float* row = sc + (size_t)m * Sn;

    float mx = -1e30f;
    for (int c = lane * 4; c < padEnd; c += 128) {
      const v4f t = *(const v4f*)(row + c);
      mx = fmaxf(mx, fmaxf(fmaxf(t[0], t[1]), fmaxf(t[2], t[3])));
    }
#pragma unroll
    for (int off = 16; off; off >>= 1) mx = fmaxf(mx, __shfl_xor(mx, off, 32));

    float sum = 0.f;
    for (int c = lane * 4; c < padEnd; c += 128) {
      v4f t = *(const v4f*)(row + c);
      t[0] = __expf(t[0] - mx); t[1] = __expf(t[1] - mx);
      t[2] = __expf(t[2] - mx); t[3] = __expf(t[3] - mx);  // masked -> 0
      *(v4f*)(row + c) = t;
      sum += t[0] + t[1] + t[2] + t[3];
    }
#pragma unroll
    for (int off = 16; off; off >>= 1) sum += __shfl_xor(sum, off, 32);
    const float inv = 1.0f / sum;

    float* wrow = wout + ((size_t)b * Sn + qBase + m) * Sn;
    for (int c = lane * 4; c < padEnd; c += 128) {
      v4f t = *(const v4f*)(row + c);
      t[0] *= inv; t[1] *= inv; t[2] *= inv; t[3] *= inv;
      *(v4f*)(row + c) = t;                               // normalized P in LDS
      __builtin_nontemporal_store(t, (v4f*)(wrow + c));   // 268 MB NT stream
    }
    const v4f z = {0.f, 0.f, 0.f, 0.f};
    for (int c = padEnd + lane * 4; c < Sn; c += 128)
      __builtin_nontemporal_store(z, (v4f*)(wrow + c));
  }
  __syncthreads();

  // ---- phase 3: output = P @ V over causal K range ----
  const int nK32    = padEnd >> 5;       // number of K=32 blocks
  const int colBase = wave * 16;         // each wave owns 16 of the 64 V cols
  const _Float16* vcol = Vb + (size_t)(colBase + ln) * Sn;  // lane's V column
  v8f o = {};
  for (int kk = 0; kk < nK32; ++kk) {
    const int k0 = kk * 32;
    // A fragment: P row ln, cols k0 + hi*8 + {0..7, 16..23} (f32 -> f16)
    const float* pp = sc + (size_t)ln * Sn + k0 + hi * 8;
    v16h pa;
#pragma unroll
    for (int i = 0; i < 8; ++i) { pa[i]     = (_Float16)pp[i];
                                  pa[8 + i] = (_Float16)pp[16 + i]; }
    // B fragment: V column colBase+ln, rows k0 + hi*8 + {0..7, 16..23}
    // -> contiguous in Vt: two 16B runs per lane
    const _Float16* vp = vcol + k0 + hi * 8;
    v16h vb;
#pragma unroll
    for (int i = 0; i < 8; ++i) { vb[i]     = vp[i];
                                  vb[8 + i] = vp[16 + i]; }
    o = __builtin_amdgcn_wmma_f32_16x16x32_f16(false, pa, false, vb,
                                               (short)0, o, false, false);
  }
#pragma unroll
  for (int r = 0; r < 8; ++r) {
    const int m = r + 8 * hi;
    out[((size_t)b * Sn + qBase + m) * DKn + colBase + ln] = o[r];
  }
}

// ---------------------------------------------------------------------------
extern "C" void kernel_launch(void* const* d_in, const int* in_sizes, int n_in,
                              void* d_out, int out_size, void* d_ws, size_t ws_size,
                              hipStream_t stream) {
  const float* x  = (const float*)d_in[0];
  // d_in[1] = mask (bool) — causality applied analytically, unused.
  const float* Wq = (const float*)d_in[2];
  const float* bq = (const float*)d_in[3];
  const float* Wk = (const float*)d_in[4];
  const float* bk = (const float*)d_in[5];
  const float* Wv = (const float*)d_in[6];
  const float* bv = (const float*)d_in[7];

  float* out_o = (float*)d_out;                      // [B,S,64]
  float* out_w = out_o + (size_t)Bn * Sn * DKn;      // [B,S,S]

  const size_t NT = (size_t)Bn * Sn * DKn;           // 1,048,576 halfs each
  _Float16* Qf = (_Float16*)d_ws;
  _Float16* Kf = Qf + NT;
  _Float16* Vt = Kf + NT;                            // [B][64][S]
  _Float16* Wt = Vt + NT;                            // [3][64][1024]

  // Weight transpose+convert: 196608 elements.
  wprep_kernel<<<dim3(196608 / 256), dim3(256), 0, stream>>>(Wq, Wk, Wv, Wt);

  // Fused QKV projection: x staged through LDS (f16), read from HBM once.
  qkv_proj_kernel<<<dim3(NTOK / 16), dim3(128), 0, stream>>>(
      x, Wt, bq, bk, bv, Qf, Kf, Vt);

  // Fused attention: 256 KB LDS score strip per workgroup.
  const int ldsBytes = 16 * Sn * (int)sizeof(float); // 262144
  (void)hipFuncSetAttribute(reinterpret_cast<const void*>(attn_kernel),
                            hipFuncAttributeMaxDynamicSharedMemorySize,
                            ldsBytes);
  attn_kernel<<<dim3(QTILES, Bn), dim3(128), ldsBytes, stream>>>(
      Qf, Kf, Vt, out_o, out_w);
}
